// MMAdaptiveNN_59210419142727
// MI455X (gfx1250) — matrix operand order repaired and checked by reading, not compiled, not run on hardware
//
#include <hip/hip_runtime.h>
#include <hip/hip_bf16.h>

typedef __attribute__((ext_vector_type(16))) __bf16 v16bf;
typedef __attribute__((ext_vector_type(8)))  float  v8f;

// ===========================================================================
// WMMA fragment layouts (CDNA5 ISA §7.12.2), wave32:
//  A (16x32 bf16): lane = half*16+lr holds row M=lr,
//      element j -> K = half*16 + j (order packed here, same as B)        (j in 0..15)
//  B (32x16 bf16): lane = half*16+lr holds col N=lr,
//      element j -> K = half*16 + j (order packed here, not the ISA table)
//  C/D (16x16 f32): VGPR r -> row M = half*8 + r, col N = lr
// Packed tile = 512 bf16 = [lane][16], 1KB, so each lane's fragment is one
// contiguous 32B vector load.
// ===========================================================================

// ---------------------------------------------------------------------------
// Pack weights: W = [W0 (Wk0 x N) ; W1] -> fragment-major bf16 tiles
// out[(nt*Ktiles + kt)*512 + lane*16 + j]
// Thread mapping is read-coalesced over columns (cc fastest).
// ---------------------------------------------------------------------------
__global__ void pack_w_kernel(const float* __restrict__ W0, int Wk0,
                              const float* __restrict__ W1,
                              int N, int Ktot,
                              __bf16* __restrict__ out, int Ntiles, int Ktiles) {
  int idx = blockIdx.x * 256 + threadIdx.x;
  int total = Ntiles * Ktiles * 512;
  if (idx >= total) return;
  int cc   = idx & 15;
  int kk   = (idx >> 4) & 31;
  int tile = idx >> 9;                 // nt*Ktiles + kt
  int nt = tile / Ktiles, kt = tile - nt * Ktiles;
  int k = kt * 32 + kk, col = nt * 16 + cc;
  float v = 0.0f;
  if (k < Ktot && col < N)
    v = (k < Wk0) ? W0[(size_t)k * N + col] : W1[(size_t)(k - Wk0) * N + col];
  int lane = ((kk >> 4) << 4) | cc;    // (kk>=16)*16 + cc
  int j = kk & 15;
  out[(size_t)tile * 512 + lane * 16 + j] = (__bf16)v;
}

// ---------------------------------------------------------------------------
// Pack activations: A(64 x Ktot) -> fragment-major bf16 tiles
// out[(mt*Ktiles + kt)*512 + lane*16 + j]
// mode 0: A = [A0 (K0) | A1-or-A1b (K1, routed by msel) | zeros]
// mode 1: A = 0.5*(A0 + A1)   (elementwise, K0 columns)
// ---------------------------------------------------------------------------
__global__ void pack_a_kernel(const float* __restrict__ A0, int K0, int lda0,
                              const float* __restrict__ A1,
                              const float* __restrict__ A1b,
                              const int* __restrict__ msel, int K1, int lda1,
                              __bf16* __restrict__ out, int Ktiles, int mode) {
  int idx = blockIdx.x * 256 + threadIdx.x;
  int total = 4 * Ktiles * 512;
  if (idx >= total) return;
  int j    = idx & 15;
  int lane = (idx >> 4) & 31;
  int tile = idx >> 9;                 // mt*Ktiles + kt
  int mt = tile / Ktiles, kt = tile - mt * Ktiles;
  int half = lane >> 4, lr = lane & 15;
  int row = mt * 16 + lr;
  int k = kt * 32 + half * 16 + j;
  float v = 0.0f;
  if (mode == 1) {
    if (k < K0)
      v = 0.5f * (A0[(size_t)row * lda0 + k] + A1[(size_t)row * lda0 + k]);
  } else {
    if (k < K0) {
      v = A0[(size_t)row * lda0 + k];
    } else {
      int k1 = k - K0;
      if (k1 < K1) {
        const float* src = (msel != nullptr && msel[row] != 0) ? A1b : A1;
        v = src[(size_t)row * lda1 + k1];
      }
    }
  }
  out[idx] = (__bf16)v;
}

// ---------------------------------------------------------------------------
// Skinny GEMM on packed fragments:
//   out(64 x N) = act( A @ W + bias ),  act: 0=none, 1=relu, 2=tanh
// One wave per 16-wide N-tile; 4 M-tiles fused so each B fragment feeds
// 4 consecutive v_wmma_f32_16x16x32_bf16. Inner loop: 5x 32B loads + 4 WMMA.
// ---------------------------------------------------------------------------
__global__ __launch_bounds__(256)
void wmma_gemm_packed(const __bf16* __restrict__ Apk,
                      const __bf16* __restrict__ Wpk,
                      const float* __restrict__ bias,
                      float* __restrict__ out, int out_stride,
                      int N, int Ntiles, int Ktiles, int act) {
  const int wid  = threadIdx.x >> 5;
  const int lane = threadIdx.x & 31;
  const int nt   = blockIdx.x * 8 + wid;
  if (nt >= Ntiles) return;

  const __bf16* wp = Wpk + (size_t)nt * Ktiles * 512 + lane * 16;
  const __bf16* ap = Apk + lane * 16;

  v8f acc0 = {}, acc1 = {}, acc2 = {}, acc3 = {};

#pragma unroll 2
  for (int kt = 0; kt < Ktiles; ++kt) {
    const v16bf bfrag = *(const v16bf*)(wp + (size_t)kt * 512);
    const v16bf a0 = *(const v16bf*)(ap + (size_t)(0 * Ktiles + kt) * 512);
    const v16bf a1 = *(const v16bf*)(ap + (size_t)(1 * Ktiles + kt) * 512);
    const v16bf a2 = *(const v16bf*)(ap + (size_t)(2 * Ktiles + kt) * 512);
    const v16bf a3 = *(const v16bf*)(ap + (size_t)(3 * Ktiles + kt) * 512);
    acc0 = __builtin_amdgcn_wmma_f32_16x16x32_bf16(false, a0, false, bfrag,
                                                   (short)0, acc0, false, false);
    acc1 = __builtin_amdgcn_wmma_f32_16x16x32_bf16(false, a1, false, bfrag,
                                                   (short)0, acc1, false, false);
    acc2 = __builtin_amdgcn_wmma_f32_16x16x32_bf16(false, a2, false, bfrag,
                                                   (short)0, acc2, false, false);
    acc3 = __builtin_amdgcn_wmma_f32_16x16x32_bf16(false, a3, false, bfrag,
                                                   (short)0, acc3, false, false);
  }

  const int half = lane >> 4;
  const int lr   = lane & 15;
  const int bcol = nt * 16 + lr;
  if (bcol < N) {
    const float bv = (bias != nullptr) ? bias[bcol] : 0.0f;
#pragma unroll
    for (int mt = 0; mt < 4; ++mt) {
      const v8f acc = (mt == 0) ? acc0 : (mt == 1) ? acc1 : (mt == 2) ? acc2 : acc3;
#pragma unroll
      for (int r = 0; r < 8; ++r) {
        const int row = mt * 16 + half * 8 + r;
        float v = acc[r] + bv;
        if (act == 1)      v = fmaxf(v, 0.0f);
        else if (act == 2) v = tanhf(v);
        out[(size_t)row * out_stride + bcol] = v;
      }
    }
  }
}

// ---------------------------------------------------------------------------
// pooled[b,d] = mean_n X[b,n,d]   (coalesced over d)
// ---------------------------------------------------------------------------
__global__ void pool_mean_kernel(const float* __restrict__ X, float* __restrict__ out,
                                 int ntok, int d, int total) {
  int i = blockIdx.x * blockDim.x + threadIdx.x;
  if (i >= total) return;
  int b = i / d, c = i % d;
  const float* p = X + (size_t)b * ntok * d + c;
  float s = 0.0f;
  for (int n = 0; n < ntok; ++n) s += p[(size_t)n * d];
  out[i] = s * (1.0f / (float)ntok);
}

// ---------------------------------------------------------------------------
// Policy head: msel = argmax(h @ W_M + b_M), l = tanh(h @ W_L + b_L)
// ---------------------------------------------------------------------------
__global__ void head_kernel(const float* __restrict__ h,
                            const float* __restrict__ W_M, const float* __restrict__ b_M,
                            const float* __restrict__ W_L, const float* __restrict__ b_L,
                            int* __restrict__ msel, float* __restrict__ l, int Dh) {
  const int b = blockIdx.x;
  const int tid = threadIdx.x;
  float a0 = 0.f, a1 = 0.f, a2 = 0.f, a3 = 0.f;
  for (int k = tid; k < Dh; k += 128) {
    const float hv = h[(size_t)b * Dh + k];
    a0 += hv * W_M[k * 2 + 0];
    a1 += hv * W_M[k * 2 + 1];
    a2 += hv * W_L[k * 2 + 0];
    a3 += hv * W_L[k * 2 + 1];
  }
  __shared__ float sm[4][128];
  sm[0][tid] = a0; sm[1][tid] = a1; sm[2][tid] = a2; sm[3][tid] = a3;
  __syncthreads();
  for (int s = 64; s > 0; s >>= 1) {
    if (tid < s)
      for (int q = 0; q < 4; ++q) sm[q][tid] += sm[q][tid + s];
    __syncthreads();
  }
  if (tid == 0) {
    const float m0 = sm[0][0] + b_M[0];
    const float m1 = sm[1][0] + b_M[1];
    msel[b] = (m1 > m0) ? 1 : 0;           // jnp.argmax: first index wins ties
    l[b * 2 + 0] = tanhf(sm[2][0] + b_L[0]);
    l[b * 2 + 1] = tanhf(sm[3][0] + b_L[1]);
  }
}

// ---------------------------------------------------------------------------
extern "C" void kernel_launch(void* const* d_in, const int* in_sizes, int n_in,
                              void* d_out, int out_size, void* d_ws, size_t ws_size,
                              hipStream_t stream) {
  (void)in_sizes; (void)n_in; (void)out_size; (void)ws_size;

  const float* X1      = (const float*)d_in[0];
  const float* X2      = (const float*)d_in[1];
  const float* z       = (const float*)d_in[2];
  const float* W_pool1 = (const float*)d_in[3];
  const float* b_pool1 = (const float*)d_in[4];
  const float* W_loc1  = (const float*)d_in[5];
  const float* b_loc1  = (const float*)d_in[6];
  const float* W_pool2 = (const float*)d_in[7];
  const float* b_pool2 = (const float*)d_in[8];
  const float* W_loc2  = (const float*)d_in[9];
  const float* b_loc2  = (const float*)d_in[10];
  const float* W_ss    = (const float*)d_in[11];
  const float* W_os    = (const float*)d_in[12];
  const float* b_s     = (const float*)d_in[13];
  const float* W_h1    = (const float*)d_in[14];
  const float* b_h1    = (const float*)d_in[15];
  const float* W_h2    = (const float*)d_in[16];
  const float* b_h2    = (const float*)d_in[17];
  const float* W_M     = (const float*)d_in[18];
  const float* b_M     = (const float*)d_in[19];
  const float* W_L     = (const float*)d_in[20];
  const float* b_L     = (const float*)d_in[21];
  // d_in[22]=W_v, d_in[23]=b_v unused in forward pass
  const float* W_q     = (const float*)d_in[24];
  const float* b_q     = (const float*)d_in[25];
  const int T = 4;

  constexpr int B = 64, DIN = 1024, D = 1024, DS = 2048, DZ = 1024, DH = 2048, C = 1000;
  // K-tile counts
  constexpr int KT1024 = 32, KT1056 = 33, KT2048 = 64, KT3072 = 96;
  // N-tile counts
  constexpr int NT1024 = 64, NT2048 = 128, NT1000 = 63;

  // ---- workspace layout ----
  float* ws = (float*)d_ws;
  float* p1   = ws;                    // 64*1024
  float* p2   = p1 + B * DIN;
  float* t1   = p2 + B * DIN;          // tanh(pool1 branch)
  float* t2   = t1 + B * D;
  float* sA   = t2 + B * D;            // 64*2048
  float* sB   = sA + B * DS;
  float* h1b  = sB + B * DS;
  float* h2b  = h1b + B * DH;
  float* ovis = h2b + B * DH;
  float* oaud = ovis + B * D;
  float* lbuf = oaud + B * D;          // 64*2
  int*   msel = (int*)(lbuf + B * 2);  // 64
  __bf16* bfw = (__bf16*)(msel + 128); // keep 32B alignment (offsets all multiples of 512 floats + 512B)

  __bf16* Wp_pool1 = bfw;                                    // 32*64*512
  __bf16* Wp_pool2 = Wp_pool1 + (size_t)KT1024 * NT1024 * 512;
  __bf16* Wp_os    = Wp_pool2 + (size_t)KT1024 * NT1024 * 512; // 32*128*512
  __bf16* Wp_h1    = Wp_os    + (size_t)KT1024 * NT2048 * 512; // 96*128*512
  __bf16* Wp_h2    = Wp_h1    + (size_t)KT3072 * NT2048 * 512; // 64*128*512
  __bf16* Wp_loc1  = Wp_h2    + (size_t)KT2048 * NT2048 * 512; // 33*64*512
  __bf16* Wp_loc2  = Wp_loc1  + (size_t)KT1056 * NT1024 * 512;
  __bf16* Wp_su    = Wp_loc2  + (size_t)KT1056 * NT1024 * 512; // 96*128*512
  __bf16* Wp_q     = Wp_su    + (size_t)KT3072 * NT2048 * 512; // 96*63*512
  __bf16* Apk      = Wp_q     + (size_t)KT3072 * NT1000 * 512; // 4*96*512 max

  auto packw = [&](const float* W0, int Wk0, const float* W1, int N, int Ktot,
                   __bf16* dst, int Ntiles, int Ktiles) {
    int total = Ntiles * Ktiles * 512;
    pack_w_kernel<<<(total + 255) / 256, 256, 0, stream>>>(W0, Wk0, W1, N, Ktot,
                                                           dst, Ntiles, Ktiles);
  };
  auto packa = [&](const float* A0, int K0, int lda0,
                   const float* A1, const float* A1b, const int* ms, int K1, int lda1,
                   int Ktiles, int mode) {
    int total = 4 * Ktiles * 512;
    pack_a_kernel<<<(total + 255) / 256, 256, 0, stream>>>(A0, K0, lda0, A1, A1b, ms,
                                                           K1, lda1, Apk, Ktiles, mode);
  };
  auto gemm = [&](const __bf16* Wpk, const float* bias, float* out, int out_stride,
                  int N, int Ntiles, int Ktiles, int act) {
    wmma_gemm_packed<<<(Ntiles + 7) / 8, 256, 0, stream>>>(Apk, Wpk, bias, out,
                                                           out_stride, N, Ntiles,
                                                           Ktiles, act);
  };

  // ---- one-time: pack all weights to fragment-major bf16 (L2-resident after) ----
  packw(W_pool1, DIN,      nullptr, D,    DIN,      Wp_pool1, NT1024, KT1024);
  packw(W_pool2, DIN,      nullptr, D,    DIN,      Wp_pool2, NT1024, KT1024);
  packw(W_os,    DIN,      nullptr, DS,   DIN,      Wp_os,    NT2048, KT1024);
  packw(W_h1,    DS + DZ,  nullptr, DH,   DS + DZ,  Wp_h1,    NT2048, KT3072);
  packw(W_h2,    DH,       nullptr, DH,   DH,       Wp_h2,    NT2048, KT2048);
  packw(W_loc1,  DIN + 2,  nullptr, D,    DIN + 2,  Wp_loc1,  NT1024, KT1056);
  packw(W_loc2,  DIN + 1,  nullptr, D,    DIN + 1,  Wp_loc2,  NT1024, KT1056);
  packw(W_ss,    DS,       W_os,    DS,   DS + D,   Wp_su,    NT2048, KT3072);
  packw(W_q,     DS + DZ,  nullptr, C,    DS + DZ,  Wp_q,     NT1000, KT3072);

  // ---- pooling (bandwidth-bound, ~156 MB) ----
  {
    const int tot = B * DIN;
    pool_mean_kernel<<<(tot + 255) / 256, 256, 0, stream>>>(X1, p1, 196, DIN, tot);
    pool_mean_kernel<<<(tot + 255) / 256, 256, 0, stream>>>(X2, p2, 400, DIN, tot);
  }

  // ---- o0 branches: t1 = tanh(p1@Wp1+b1), t2 = tanh(p2@Wp2+b2) ----
  packa(p1, DIN, DIN, nullptr, nullptr, nullptr, 0, 0, KT1024, 0);
  gemm(Wp_pool1, b_pool1, t1, D, D, NT1024, KT1024, /*tanh*/2);
  packa(p2, DIN, DIN, nullptr, nullptr, nullptr, 0, 0, KT1024, 0);
  gemm(Wp_pool2, b_pool2, t2, D, D, NT1024, KT1024, /*tanh*/2);

  // ---- s = tanh(o0 @ W_os + b_s), o0 = 0.5*(t1+t2) folded into the pack ----
  packa(t1, D, D, t2, nullptr, nullptr, 0, 0, KT1024, /*avg*/1);
  gemm(Wp_os, b_s, sA, DS, DS, NT2048, KT1024, /*tanh*/2);

  float* cur = sA;
  float* nxt = sB;
  float* y = (float*)d_out;  // (B, T, C)

  for (int t = 0; t < T; ++t) {
    // h1 = relu([s, z] @ W_h1 + b_h1)
    packa(cur, DS, DS, z, nullptr, nullptr, DZ, DZ, KT3072, 0);
    gemm(Wp_h1, b_h1, h1b, DH, DH, NT2048, KT3072, /*relu*/1);
    // h2 = relu(h1 @ W_h2 + b_h2)
    packa(h1b, DH, DH, nullptr, nullptr, nullptr, 0, 0, KT2048, 0);
    gemm(Wp_h2, b_h2, h2b, DH, DH, NT2048, KT2048, /*relu*/1);
    // modality argmax + location tanh
    head_kernel<<<B, 128, 0, stream>>>(h2b, W_M, b_M, W_L, b_L, msel, lbuf, DH);
    // o_vis = tanh([pooled1, l] @ W_loc1 + b_loc1), K = 1026 padded to 1056
    packa(p1, DIN, DIN, lbuf, nullptr, nullptr, 2, 2, KT1056, 0);
    gemm(Wp_loc1, b_loc1, ovis, D, D, NT1024, KT1056, /*tanh*/2);
    // o_aud = tanh([pooled2, l[:, :1]] @ W_loc2 + b_loc2), K = 1025 padded
    packa(p2, DIN, DIN, lbuf, nullptr, nullptr, 1, 2, KT1056, 0);
    gemm(Wp_loc2, b_loc2, oaud, D, D, NT1024, KT1056, /*tanh*/2);
    // s' = tanh([s, o_t] @ [W_ss; W_os] + b_s), o_t routed per-row by msel
    packa(cur, DS, DS, ovis, oaud, msel, D, D, KT3072, 0);
    gemm(Wp_su, b_s, nxt, DS, DS, NT2048, KT3072, /*tanh*/2);
    // y_t = [s', z] @ W_q + b_q  (N=1000 tail via col guard)
    packa(nxt, DS, DS, z, nullptr, nullptr, DZ, DZ, KT3072, 0);
    gemm(Wp_q, b_q, y + (size_t)t * C, T * C, C, NT1000, KT3072, /*none*/0);

    float* tmp = cur; cur = nxt; nxt = tmp;
  }
}
